// Attention_39573828665669
// MI455X (gfx1250) — compile-verified, matrix operand order checked by reading, not run on hardware
//
#include <hip/hip_runtime.h>
#include <hip/hip_bf16.h>

// MI455X / gfx1250 fused MHA:
//   1) Q/K/V projections: fp32 GEMM -> bf16 activations, v_wmma_f32_16x16x32_bf16
//   2) flash attention (online softmax) with position_bias + mask, all WMMA
//   3) output projection -> fp32 d_out
// Workspace layout (bf16 = raw ushort bits):
//   Q  : [H][S][DH]   at ws + 0
//   K  : [H][S][DH]   at ws + 8MB
//   Vt : [H][DH][S]   at ws + 16MB
//   Ctx: [S][H*DH]    at ws + 24MB          (total 32 MB required)

#define S_LEN 2048
#define DMODEL 2048
#define NHEAD 32
#define DHEAD 64

typedef __attribute__((ext_vector_type(16))) __bf16 bf16x16;
typedef __attribute__((ext_vector_type(8)))  float  f32x8;
typedef __attribute__((__vector_size__(4 * sizeof(int)))) int v4i;

// ---- gfx1250 async global->LDS copies (ASYNCcnt path), guarded so a missing
// ---- builtin falls back to plain load/ds_store staging.
#if defined(__gfx1250__) && __has_builtin(__builtin_amdgcn_global_load_async_to_lds_b128) && \
    __has_builtin(__builtin_amdgcn_s_wait_asynccnt)
#define HAVE_ASYNC_LDS 1
#else
#define HAVE_ASYNC_LDS 0
#endif

// typed address-space casts matching the builtin's signature (v4i pointees)
#define ASG(p) ((__attribute__((address_space(1))) v4i*)(unsigned long long)(const void*)(p))
#define ASL(p) ((__attribute__((address_space(3))) v4i*)(unsigned int)(unsigned long long)(void*)(p))

union Frag {
  bf16x16 v;
  uint4   q[2];   // two 16B chunks: elements 0..7 and 8..15
};

__device__ __forceinline__ unsigned short f2bf(float f) {
  unsigned int u = __float_as_uint(f);
  u += 0x7fffu + ((u >> 16) & 1u);          // round-to-nearest-even
  return (unsigned short)(u >> 16);
}

// pack two fp32 -> two bf16 in one VALU op when v_cvt_pk_bf16_f32 is available
__device__ __forceinline__ unsigned int pack2bf(float lo, float hi) {
#if defined(__gfx1250__) && __has_builtin(__builtin_amdgcn_cvt_pk_bf16_f32)
  auto t = __builtin_amdgcn_cvt_pk_bf16_f32(lo, hi);
  unsigned int u;
  __builtin_memcpy(&u, &t, 4);
  return u;
#else
  return ((unsigned int)f2bf(hi) << 16) | (unsigned int)f2bf(lo);
#endif
}

__device__ __forceinline__ f32x8 zero8() {
  f32x8 z;
#pragma unroll
  for (int i = 0; i < 8; ++i) z[i] = 0.0f;
  return z;
}

__device__ __forceinline__ f32x8 wmma_bf16(const Frag& a, const Frag& b, f32x8 c) {
  return __builtin_amdgcn_wmma_f32_16x16x32_bf16(false, a.v, false, b.v,
                                                 (short)0, c, false, false);
}

// ---------------------------------------------------------------------------
// Generic GEMM:  Out[s][n] = sum_k A[s][k] * W[n][k] + bias[n]
// M = N = K = 2048. Block = 256 threads (8 waves), 128x128 output tile.
// OUTL: 0 = bf16 [H][S][DH] (Q/K), 1 = bf16 [H][DH][S] (V^T), 2 = f32 [S][DM]
// ---------------------------------------------------------------------------
template <typename TA, int OUTL>
__global__ __launch_bounds__(256) void gemm_bias(const TA*    __restrict__ A,
                                                 const float* __restrict__ W,
                                                 const float* __restrict__ bias,
                                                 void*        __restrict__ Out) {
  __shared__ __align__(16) unsigned short As[128 * 32];
  __shared__ __align__(16) unsigned short Bs[128 * 32];

  const int tid  = threadIdx.x;
  const int lane = tid & 31;
  const int hl   = (lane >> 4) & 1;
  const int ln16 = lane & 15;
  const int w    = tid >> 5;     // wave 0..7
  const int wr   = w >> 2;       // 0..1  -> 64 rows
  const int wc   = w & 3;        // 0..3  -> 32 cols
  const int rowBase = blockIdx.x * 128;
  const int colBase = blockIdx.y * 128;

  f32x8 acc[4][2];
#pragma unroll
  for (int i = 0; i < 4; ++i)
#pragma unroll
    for (int j = 0; j < 2; ++j) acc[i][j] = zero8();

  for (int k0 = 0; k0 < DMODEL; k0 += 32) {
    __syncthreads();
    // ---- stage A tile 128x32
    if constexpr (sizeof(TA) == 2) {
      // A already bf16: raw copy, 16B chunks (async DMA path when available)
#pragma unroll
      for (int it = 0; it < 2; ++it) {
        int idx = tid + it * 256;        // 512 chunks of 8 elements
        int r = idx >> 2;
        int c = (idx & 3) * 8;
        const unsigned short* src = (const unsigned short*)A + (size_t)(rowBase + r) * DMODEL + k0 + c;
#if HAVE_ASYNC_LDS
        __builtin_amdgcn_global_load_async_to_lds_b128(ASG(src), ASL(&As[r * 32 + c]), 0, 0);
#else
        *(uint4*)&As[r * 32 + c] = *(const uint4*)src;
#endif
      }
    } else {
      // A fp32: convert to bf16 with packed converts
#pragma unroll
      for (int it = 0; it < 4; ++it) {
        int idx = tid + it * 256;        // 1024 chunks of 4 elements
        int r = idx >> 3;
        int c = (idx & 7) * 4;
        const float* src = (const float*)A + (size_t)(rowBase + r) * DMODEL + k0 + c;
        uint2 d;
        d.x = pack2bf(src[0], src[1]);
        d.y = pack2bf(src[2], src[3]);
        *(uint2*)&As[r * 32 + c] = d;
      }
    }
    // ---- stage B tile 128x32 from W[n][k] (always fp32 -> bf16)
#pragma unroll
    for (int it = 0; it < 4; ++it) {
      int idx = tid + it * 256;
      int r = idx >> 3;
      int c = (idx & 7) * 4;
      const float* src = W + (size_t)(colBase + r) * DMODEL + k0 + c;
      uint2 d;
      d.x = pack2bf(src[0], src[1]);
      d.y = pack2bf(src[2], src[3]);
      *(uint2*)&Bs[r * 32 + c] = d;
    }
#if HAVE_ASYNC_LDS
    if constexpr (sizeof(TA) == 2) __builtin_amdgcn_s_wait_asynccnt(0);
#endif
    __syncthreads();

    Frag af[4], bfv[2];
#pragma unroll
    for (int ti = 0; ti < 4; ++ti) {
      int r = wr * 64 + ti * 16 + ln16;
      af[ti].q[0] = *(const uint4*)&As[r * 32 + 8 * hl];
      af[ti].q[1] = *(const uint4*)&As[r * 32 + 16 + 8 * hl];
    }
#pragma unroll
    for (int tj = 0; tj < 2; ++tj) {
      int r = wc * 32 + tj * 16 + ln16;
      bfv[tj].q[0] = *(const uint4*)&Bs[r * 32 + 8 * hl];
      bfv[tj].q[1] = *(const uint4*)&Bs[r * 32 + 16 + 8 * hl];
    }
#pragma unroll
    for (int ti = 0; ti < 4; ++ti)
#pragma unroll
      for (int tj = 0; tj < 2; ++tj)
        acc[ti][tj] = wmma_bf16(af[ti], bfv[tj], acc[ti][tj]);
  }

  // epilogue: bias add + layout-specific store
#pragma unroll
  for (int ti = 0; ti < 4; ++ti) {
#pragma unroll
    for (int tj = 0; tj < 2; ++tj) {
      int n = colBase + wc * 32 + tj * 16 + ln16;
      float bn = bias[n];
#pragma unroll
      for (int r = 0; r < 8; ++r) {
        int srow = rowBase + wr * 64 + ti * 16 + r + 8 * hl;   // C/D: M = r + 8*hl
        float val = acc[ti][tj][r] + bn;
        if constexpr (OUTL == 0) {
          ((unsigned short*)Out)[(size_t)(n >> 6) * S_LEN * DHEAD +
                                 (size_t)srow * DHEAD + (n & 63)] = f2bf(val);
        } else if constexpr (OUTL == 1) {
          ((unsigned short*)Out)[(size_t)(n >> 6) * DHEAD * S_LEN +
                                 (size_t)(n & 63) * S_LEN + srow] = f2bf(val);
        } else {
          ((float*)Out)[(size_t)srow * DMODEL + n] = val;
        }
      }
    }
  }
}

// ---------------------------------------------------------------------------
// Flash attention: grid (S/128, H), 256 threads. Each wave owns 16 q rows.
// K/V 64-wide tiles staged in LDS (async DMA when available), scores + PV via
// bf16 WMMA, online softmax, position_bias streamed with prefetch.
// ---------------------------------------------------------------------------
__global__ __launch_bounds__(256) void attn_kernel(const unsigned short* __restrict__ Q,
                                                   const unsigned short* __restrict__ K,
                                                   const unsigned short* __restrict__ Vt,
                                                   const unsigned char*  __restrict__ mask,
                                                   const float*          __restrict__ pb,
                                                   unsigned short*       __restrict__ Ctx) {
  __shared__ __align__(16) unsigned short Ks[64 * 64];
  __shared__ __align__(16) unsigned short Vs[64 * 64];
  __shared__ __align__(16) unsigned short Ps[8][16 * 64];

  const int tid  = threadIdx.x;
  const int lane = tid & 31;
  const int hl   = (lane >> 4) & 1;
  const int ln16 = lane & 15;
  const int w    = tid >> 5;
  const int h    = blockIdx.y;
  const int qBase = blockIdx.x * 128 + w * 16;

  const unsigned short* Qh  = Q  + (size_t)h * S_LEN * DHEAD;
  const unsigned short* Kh  = K  + (size_t)h * S_LEN * DHEAD;
  const unsigned short* Vh  = Vt + (size_t)h * DHEAD * S_LEN;
  const float*          pbh = pb + (size_t)h * S_LEN * S_LEN;

  // Q A-fragments (16 rows x 64 d) resident in registers for the whole pass
  Frag aq[2];
#pragma unroll
  for (int kk = 0; kk < 2; ++kk) {
    const unsigned short* row = Qh + (size_t)(qBase + ln16) * DHEAD + 32 * kk + 8 * hl;
    aq[kk].q[0] = *(const uint4*)row;
    aq[kk].q[1] = *(const uint4*)(row + 16);
  }

  float mrow[8], lrow[8];
  f32x8 oacc[4];
#pragma unroll
  for (int r = 0; r < 8; ++r) { mrow[r] = -INFINITY; lrow[r] = 0.0f; }
#pragma unroll
  for (int j = 0; j < 4; ++j) oacc[j] = zero8();

  const float sm_scale = 0.125f;   // 1/sqrt(DH)

  for (int kv = 0; kv < S_LEN; kv += 64) {
    __syncthreads();
    // cooperative stage: K tile [kv][d], V^T tile [d][kv] (8KB each)
#pragma unroll
    for (int it = 0; it < 2; ++it) {
      int idx = tid + it * 256;          // 512 chunks of 8 bf16
      int r = idx >> 3;
      int c = (idx & 7) * 8;
      const unsigned short* ksrc = Kh + (size_t)(kv + r) * DHEAD + c;
      const unsigned short* vsrc = Vh + (size_t)r * S_LEN + kv + c;
#if HAVE_ASYNC_LDS
      __builtin_amdgcn_global_load_async_to_lds_b128(ASG(ksrc), ASL(&Ks[r * 64 + c]), 0, 0);
      __builtin_amdgcn_global_load_async_to_lds_b128(ASG(vsrc), ASL(&Vs[r * 64 + c]), 0, 0);
#else
      *(uint4*)&Ks[r * 64 + c] = *(const uint4*)ksrc;
      *(uint4*)&Vs[r * 64 + c] = *(const uint4*)vsrc;
#endif
    }
    // keep the dominant position_bias stream (512MB) ahead of the compute loop
    if (kv + 64 < S_LEN) {
      __builtin_prefetch(pbh + (size_t)(qBase + ln16) * S_LEN + (kv + 64) + hl * 32, 0, 0);
    }
#if HAVE_ASYNC_LDS
    __builtin_amdgcn_s_wait_asynccnt(0);
#endif
    __syncthreads();

    // scores: 16 q rows x 64 kv cols
    f32x8 sacc[4];
#pragma unroll
    for (int j = 0; j < 4; ++j) sacc[j] = zero8();
#pragma unroll
    for (int kk = 0; kk < 2; ++kk) {
#pragma unroll
      for (int j = 0; j < 4; ++j) {
        Frag bk;
        int r = j * 16 + ln16;                       // kv row, B-frag N = lane%16
        bk.q[0] = *(const uint4*)&Ks[r * 64 + 32 * kk + 8 * hl];
        bk.q[1] = *(const uint4*)&Ks[r * 64 + 32 * kk + 16 + 8 * hl];
        sacc[j] = wmma_bf16(aq[kk], bk, sacc[j]);
      }
    }

    // scale + position_bias stream + mask (D layout: row = r+8*hl, col = 16j+ln16)
#pragma unroll
    for (int j = 0; j < 4; ++j) {
      int col = kv + j * 16 + ln16;
#pragma unroll
      for (int r = 0; r < 8; ++r) {
        int qrow = qBase + r + 8 * hl;
        float s = sacc[j][r] * sm_scale + pbh[(size_t)qrow * S_LEN + col];
        s = mask[(size_t)qrow * S_LEN + col] ? s : -INFINITY;
        sacc[j][r] = s;
      }
    }

    // online softmax: row max via shfl within 16-lane half (keeps hl group)
    float mnew[8], sc[8];
#pragma unroll
    for (int r = 0; r < 8; ++r) {
      float mx = sacc[0][r];
#pragma unroll
      for (int j = 1; j < 4; ++j) mx = fmaxf(mx, sacc[j][r]);
#pragma unroll
      for (int d = 1; d < 16; d <<= 1) mx = fmaxf(mx, __shfl_xor(mx, d, 32));
      mnew[r] = fmaxf(mrow[r], mx);
      sc[r] = (mrow[r] == -INFINITY) ? 0.0f : __expf(mrow[r] - mnew[r]);
      mrow[r] = mnew[r];
    }

    // p = exp(s - m); spill P to per-wave LDS patch (D layout -> A layout bridge)
    float rs[8];
#pragma unroll
    for (int r = 0; r < 8; ++r) rs[r] = 0.0f;
#pragma unroll
    for (int j = 0; j < 4; ++j) {
#pragma unroll
      for (int r = 0; r < 8; ++r) {
        float s = sacc[j][r];
        float p = (s == -INFINITY) ? 0.0f : __expf(s - mnew[r]);
        rs[r] += p;
        Ps[w][(r + 8 * hl) * 64 + j * 16 + ln16] = f2bf(p);
      }
    }
#pragma unroll
    for (int r = 0; r < 8; ++r) {
      float t = rs[r];
#pragma unroll
      for (int d = 1; d < 16; d <<= 1) t += __shfl_xor(t, d, 32);
      lrow[r] = lrow[r] * sc[r] + t;
    }

    // rescale O accumulators by per-row correction
#pragma unroll
    for (int j = 0; j < 4; ++j)
#pragma unroll
      for (int r = 0; r < 8; ++r) oacc[j][r] *= sc[r];

    // O += P @ V  (A = P 16x64 from LDS, B = V^T tile rows = d)
    Frag ap[2];
#pragma unroll
    for (int kk = 0; kk < 2; ++kk) {
      ap[kk].q[0] = *(const uint4*)&Ps[w][ln16 * 64 + 32 * kk + 8 * hl];
      ap[kk].q[1] = *(const uint4*)&Ps[w][ln16 * 64 + 32 * kk + 16 + 8 * hl];
    }
#pragma unroll
    for (int kk = 0; kk < 2; ++kk) {
#pragma unroll
      for (int j = 0; j < 4; ++j) {
        Frag bv;
        int r = j * 16 + ln16;                       // d row in Vs
        bv.q[0] = *(const uint4*)&Vs[r * 64 + 32 * kk + 8 * hl];
        bv.q[1] = *(const uint4*)&Vs[r * 64 + 32 * kk + 16 + 8 * hl];
        oacc[j] = wmma_bf16(ap[kk], bv, oacc[j]);
      }
    }
  }

  // normalize, write context bf16 as [S][H*DH]
#pragma unroll
  for (int r = 0; r < 8; ++r) {
    int qrow = qBase + r + 8 * hl;
    float inv = (lrow[r] > 0.0f) ? (1.0f / lrow[r]) : 0.0f;
#pragma unroll
    for (int j = 0; j < 4; ++j) {
      int d = j * 16 + ln16;
      Ctx[(size_t)qrow * (NHEAD * DHEAD) + h * DHEAD + d] = f2bf(oacc[j][r] * inv);
    }
  }
}

// ---------------------------------------------------------------------------
extern "C" void kernel_launch(void* const* d_in, const int* in_sizes, int n_in,
                              void* d_out, int out_size, void* d_ws, size_t ws_size,
                              hipStream_t stream) {
  const float*         hq   = (const float*)d_in[0];
  const float*         hkv  = (const float*)d_in[1];
  const unsigned char* mask = (const unsigned char*)d_in[2];   // jax bool = 1 byte
  const float*         pb   = (const float*)d_in[3];
  const float*         wq   = (const float*)d_in[4];
  const float*         bq   = (const float*)d_in[5];
  const float*         wk   = (const float*)d_in[6];
  const float*         bk   = (const float*)d_in[7];
  const float*         wv   = (const float*)d_in[8];
  const float*         bv   = (const float*)d_in[9];
  const float*         wo   = (const float*)d_in[10];
  const float*         bo   = (const float*)d_in[11];

  const size_t NELT = (size_t)S_LEN * DMODEL;                  // 4M elements
  unsigned short* Qbf = (unsigned short*)d_ws;
  unsigned short* Kbf = Qbf + NELT;
  unsigned short* Vtb = Kbf + NELT;
  unsigned short* Ctx = Vtb + NELT;                            // needs 32MB total

  dim3 gridG(DMODEL / 128, DMODEL / 128);                      // 16 x 16
  dim3 blk(256);

  gemm_bias<float, 0><<<gridG, blk, 0, stream>>>(hq,  wq, bq, (void*)Qbf);
  gemm_bias<float, 0><<<gridG, blk, 0, stream>>>(hkv, wk, bk, (void*)Kbf);
  gemm_bias<float, 1><<<gridG, blk, 0, stream>>>(hkv, wv, bv, (void*)Vtb);

  dim3 gridA(S_LEN / 128, NHEAD);                              // 16 x 32
  attn_kernel<<<gridA, blk, 0, stream>>>(Qbf, Kbf, Vtb, mask, pb, Ctx);

  gemm_bias<unsigned short, 2><<<gridG, blk, 0, stream>>>(Ctx, wo, bo, d_out);
}
